// RPEAttention_85375359910652
// MI455X (gfx1250) — matrix operand order repaired and checked by reading, not compiled, not run on hardware
//
#include <hip/hip_runtime.h>
#include <hip/hip_bf16.h>

typedef __attribute__((ext_vector_type(16))) _Float16 v16h;
typedef __attribute__((ext_vector_type(8)))  _Float16 v8h;
typedef __attribute__((ext_vector_type(8)))  float    v8f;

union AFrag { v16h v; struct { v8h lo, hi; } p; };

#define WMMA_F16(A, B, C) \
  __builtin_amdgcn_wmma_f32_16x16x32_f16(false, (A), false, (B), (short)0, (C), false, false)

static __device__ __forceinline__ int imin(int a, int b) { return a < b ? a : b; }

// ---------------------------------------------------------------------------
// Problem constants
// ---------------------------------------------------------------------------
#define B_  16
#define T_  500
#define C_  512
#define H_  8
#define HD_ 64
#define NQKV_ 1536
#define NB_ 49
#define ROWS_ (B_ * T_)       // 8000

// ---------------------------------------------------------------------------
// Pre-pass: f32 -> f16 copy / transposed copy
// ---------------------------------------------------------------------------
__global__ void cvt_f16_kernel(const float* __restrict__ src, _Float16* __restrict__ dst, int n) {
  int i = blockIdx.x * blockDim.x + threadIdx.x;
  if (i < n) dst[i] = (_Float16)src[i];
}

// W[k][n] (K x N row-major)  ->  WT[n][k] (N x K row-major, k contiguous)
__global__ void transpose_f16_kernel(const float* __restrict__ W, _Float16* __restrict__ WT,
                                     int K, int N) {
  int i = blockIdx.x * blockDim.x + threadIdx.x;
  if (i < K * N) {
    int n = i / K, k = i - n * K;
    WT[i] = (_Float16)W[(size_t)k * N + n];
  }
}

// ---------------------------------------------------------------------------
// GEMM1: xh (8000x512) @ W_attn (512x1536) + b_attn, scattered as f16 into
// Q/K/V laid out [B*H][T][64]. Q is pre-scaled by 1/sqrt(hd)=0.125.
// One wave computes a 32x64 output tile: 2 A-fragments share each B-fragment
// (8 WMMAs per k-step, halves B traffic per FLOP).
// ---------------------------------------------------------------------------
__global__ __launch_bounds__(256) void gemm_qkv_kernel(
    const _Float16* __restrict__ xh, const _Float16* __restrict__ WT,
    const float* __restrict__ b_attn,
    _Float16* __restrict__ Qh, _Float16* __restrict__ Kh, _Float16* __restrict__ Vh) {
  const int lane = threadIdx.x & 31;
  const int wave = blockIdx.x * (blockDim.x >> 5) + (threadIdx.x >> 5);
  const int NT = NQKV_ / 64;                    // 24
  if (wave >= (ROWS_ / 32) * NT) return;
  const int mt = wave / NT, nt = wave - mt * NT;
  const int ln = lane & 15, hi = lane >> 4;

  const _Float16* arow0 = xh + (size_t)(mt * 32 + ln) * C_ + hi * 16;
  const _Float16* arow1 = arow0 + (size_t)16 * C_;
  const _Float16* brow[4];
  int ncol[4];
  v8f acc0[4] = {}, acc1[4] = {};
#pragma unroll
  for (int j = 0; j < 4; ++j) {
    ncol[j] = nt * 64 + 16 * j + ln;
    brow[j] = WT + (size_t)ncol[j] * C_ + hi * 16;
  }

  for (int k0 = 0; k0 < C_; k0 += 32) {
    __builtin_prefetch((const void*)(arow0 + k0 + 32), 0, 0);  // global_prefetch_b8
    AFrag a0, a1;
    a0.p.lo = *(const v8h*)(arow0 + k0);
    a0.p.hi = *(const v8h*)(arow0 + k0 + 8);
    a1.p.lo = *(const v8h*)(arow1 + k0);
    a1.p.hi = *(const v8h*)(arow1 + k0 + 8);
#pragma unroll
    for (int j = 0; j < 4; ++j) {
      v16h b = *(const v16h*)(brow[j] + k0);
      acc0[j] = WMMA_F16(a0.v, b, acc0[j]);
      acc1[j] = WMMA_F16(a1.v, b, acc1[j]);
    }
  }

#pragma unroll
  for (int j = 0; j < 4; ++j) {
    const int col = ncol[j];
    const float bias = b_attn[col];
    const int sec = col >> 9;                   // 0=q 1=k 2=v
    const int c = col & 511, h = c >> 6, d = c & 63;
    _Float16* dst = (sec == 0) ? Qh : (sec == 1) ? Kh : Vh;
    const float scale = (sec == 0) ? 0.125f : 1.0f;
#pragma unroll
    for (int i = 0; i < 2; ++i) {
      const v8f* acc = (i == 0) ? acc0 : acc1;
#pragma unroll
      for (int r = 0; r < 8; ++r) {
        const int grow = mt * 32 + i * 16 + r + hi * 8;    // 0..7999
        const int b_ = grow / T_, t = grow - b_ * T_;
        const float v = (acc[j][r] + bias) * scale;
        dst[(((size_t)(b_ * H_ + h)) * T_ + t) * HD_ + d] = (_Float16)v;
      }
    }
  }
}

// ---------------------------------------------------------------------------
// Fused flash attention with RPE bucket bias.
// 4 waves / block, one (b, h, 16-row query tile) per wave; per-wave LDS
// regions; all waves in a block share (b,h) so the LUT stage is amortized.
// Streams K/V in 32-column steps; online softmax; P goes through LDS
// (C-layout -> A-layout); V staged transposed in LDS so B-fragments are
// contiguous ds reads.
// Faithful to the reference raw-view: lut row = b>>1, ct batch = (b&1)*8+h.
// ---------------------------------------------------------------------------
__global__ __launch_bounds__(128) void attn_kernel(
    const _Float16* __restrict__ Qh, const _Float16* __restrict__ Kh,
    const _Float16* __restrict__ Vh, const float* __restrict__ ct,
    const float* __restrict__ lut, _Float16* __restrict__ Yh) {
  __shared__ __align__(64) _Float16 PsAll[4][16][32];
  __shared__ __align__(64) _Float16 VtAll[4][64][32];
  __shared__ float lutr[NB_];

  const int lane = threadIdx.x & 31;
  const int wid = threadIdx.x >> 5;
  const int bh = blockIdx.x >> 3;               // 0..127
  const int tgroup = blockIdx.x & 7;
  const int ttile = tgroup * 4 + wid;           // 0..31
  const int b = bh >> 3, h = bh & 7;
  const int t0 = ttile * 16;
  const int ln = lane & 15, hi = lane >> 4;

  _Float16 (*Ps)[32] = PsAll[wid];
  _Float16 (*Vt)[32] = VtAll[wid];

  for (int i = threadIdx.x; i < NB_; i += 128) lutr[i] = lut[(b >> 1) * NB_ + i];
  __syncthreads();

  const _Float16* Qb = Qh + (size_t)bh * T_ * HD_;
  const _Float16* Kb = Kh + (size_t)bh * T_ * HD_;
  const _Float16* Vb = Vh + (size_t)bh * T_ * HD_;
  const int bct = (b & 1) * 8 + h;                       // quirky raw-view batch
  const float2* ctb = (const float2*)ct + (size_t)bct * T_;

  // Q A-fragments (K = 0..31 and 32..63)
  AFrag aQ0, aQ1;
  {
    const _Float16* qr = Qb + (size_t)imin(t0 + ln, T_ - 1) * HD_ + hi * 16;
    aQ0.p.lo = *(const v8h*)(qr);
    aQ0.p.hi = *(const v8h*)(qr + 8);
    aQ1.p.lo = *(const v8h*)(qr + 32);
    aQ1.p.hi = *(const v8h*)(qr + 40);
  }

  float qx[8], qy[8], mrow[8], lrow[8];
  v8f acc[4] = {};
#pragma unroll
  for (int r = 0; r < 8; ++r) {
    float2 cq = ctb[imin(t0 + r + hi * 8, T_ - 1)];
    qx[r] = cq.x; qy[r] = cq.y;
    mrow[r] = -__builtin_inff();
    lrow[r] = 0.0f;
  }

  const float INV_A = 1.0f / 12.0f;
  const float LOG_SCALE = 12.0f / 2.0794415416798357f;   // (beta-alpha)/log(gamma/alpha)

  for (int s0 = 0; s0 < 512; s0 += 32) {
    const int sc1 = s0 + 16 + ln;
    const int sa0 = imin(s0 + ln, T_ - 1), sa1 = imin(sc1, T_ - 1);

    // ---- scores: two 16x16 WMMA C-tiles, K=64 chained over two steps ----
    v8f c0 = {}, c1 = {};
    {
      const _Float16* kr0 = Kb + (size_t)sa0 * HD_ + hi * 16;
      const _Float16* kr1 = Kb + (size_t)sa1 * HD_ + hi * 16;
      v16h b00 = *(const v16h*)(kr0);
      v16h b01 = *(const v16h*)(kr0 + 32);
      v16h b10 = *(const v16h*)(kr1);
      v16h b11 = *(const v16h*)(kr1 + 32);
      c0 = WMMA_F16(aQ0.v, b00, c0);
      c0 = WMMA_F16(aQ1.v, b01, c0);
      c1 = WMMA_F16(aQ0.v, b10, c1);
      c1 = WMMA_F16(aQ1.v, b11, c1);
    }

    // ---- RPE bias + mask ----
    const float2 cs0 = ctb[sa0], cs1 = ctb[sa1];
    const bool msk1 = (sc1 >= T_);                       // col0 is always < 500
    float tmax[8], p0[8], p1[8], tsum[8];
#pragma unroll
    for (int r = 0; r < 8; ++r) {
      float dx0 = qx[r] - cs0.x, dy0 = qy[r] - cs0.y;
      float dx1 = qx[r] - cs1.x, dy1 = qy[r] - cs1.y;
      float dist0 = truncf(__fsqrt_rn(dx0 * dx0 + dy0 * dy0) * INV_A);
      float dist1 = truncf(__fsqrt_rn(dx1 * dx1 + dy1 * dy1) * INV_A);
      float v0 = (dist0 <= 12.0f) ? dist0
               : fminf(24.0f, rintf(12.0f + __logf(dist0 * INV_A) * LOG_SCALE));
      float v1 = (dist1 <= 12.0f) ? dist1
               : fminf(24.0f, rintf(12.0f + __logf(dist1 * INV_A) * LOG_SCALE));
      float e0 = c0[r] + lutr[(int)v0 + 24];
      float e1 = c1[r] + lutr[(int)v1 + 24];
      if (msk1) e1 = -__builtin_inff();
      c0[r] = e0; c1[r] = e1;
      tmax[r] = fmaxf(e0, e1);
    }
    // row max across the 16 lanes of each half
#pragma unroll
    for (int off = 1; off < 16; off <<= 1)
#pragma unroll
      for (int r = 0; r < 8; ++r)
        tmax[r] = fmaxf(tmax[r], __shfl_xor(tmax[r], off, 32));

#pragma unroll
    for (int r = 0; r < 8; ++r) {
      const float mn = fmaxf(mrow[r], tmax[r]);
      const float corr = __expf(mrow[r] - mn);
      mrow[r] = mn;
      p0[r] = __expf(c0[r] - mn);
      p1[r] = __expf(c1[r] - mn);
      tsum[r] = p0[r] + p1[r];
      lrow[r] *= corr;
#pragma unroll
      for (int j = 0; j < 4; ++j) acc[j][r] = acc[j][r] * corr;
    }
#pragma unroll
    for (int off = 1; off < 16; off <<= 1)
#pragma unroll
      for (int r = 0; r < 8; ++r)
        tsum[r] += __shfl_xor(tsum[r], off, 32);
#pragma unroll
    for (int r = 0; r < 8; ++r) lrow[r] += tsum[r];

    // ---- P: C-layout -> LDS -> A-layout f16 fragment ----
#pragma unroll
    for (int r = 0; r < 8; ++r) {
      Ps[r + hi * 8][ln]      = (_Float16)p0[r];
      Ps[r + hi * 8][16 + ln] = (_Float16)p1[r];
    }
    // ---- V tile staged transposed: Vt[d][s-s0] ----
    {
      const _Float16* vr = Vb + (size_t)imin(s0 + lane, T_ - 1) * HD_;
#pragma unroll
      for (int d8 = 0; d8 < HD_; d8 += 8) {
        v8h vv = *(const v8h*)(vr + d8);
#pragma unroll
        for (int e = 0; e < 8; ++e) Vt[d8 + e][lane] = vv[e];
      }
    }
    __syncthreads();

    AFrag aP;
    {
      const _Float16* pr = &Ps[ln][hi * 16];
      aP.p.lo = *(const v8h*)(pr);
      aP.p.hi = *(const v8h*)(pr + 8);
    }
#pragma unroll
    for (int j = 0; j < 4; ++j) {
      v16h bV = *(const v16h*)(&Vt[16 * j + ln][hi * 16]);
      acc[j] = WMMA_F16(aP.v, bV, acc[j]);
    }
    __syncthreads();
  }

  // ---- normalize and write y into [B*T][512] f16 (ready for out-proj) ----
#pragma unroll
  for (int j = 0; j < 4; ++j)
#pragma unroll
    for (int r = 0; r < 8; ++r) {
      const int t = t0 + r + hi * 8;
      if (t < T_) {
        const float y = acc[j][r] / lrow[r];
        Yh[((size_t)(b * T_ + t)) * C_ + h * HD_ + 16 * j + ln] = (_Float16)y;
      }
    }
}

// ---------------------------------------------------------------------------
// GEMM2: Yh (8000x512) @ W_proj (512x512) + b_proj -> out f32
// 32x64 tile per wave, same structure as GEMM1.
// ---------------------------------------------------------------------------
__global__ __launch_bounds__(256) void gemm_proj_kernel(
    const _Float16* __restrict__ Yh, const _Float16* __restrict__ WT,
    const float* __restrict__ b_proj, float* __restrict__ out) {
  const int lane = threadIdx.x & 31;
  const int wave = blockIdx.x * (blockDim.x >> 5) + (threadIdx.x >> 5);
  const int NT = C_ / 64;                       // 8
  if (wave >= (ROWS_ / 32) * NT) return;
  const int mt = wave / NT, nt = wave - mt * NT;
  const int ln = lane & 15, hi = lane >> 4;

  const _Float16* arow0 = Yh + (size_t)(mt * 32 + ln) * C_ + hi * 16;
  const _Float16* arow1 = arow0 + (size_t)16 * C_;
  const _Float16* brow[4];
  int ncol[4];
  v8f acc0[4] = {}, acc1[4] = {};
#pragma unroll
  for (int j = 0; j < 4; ++j) {
    ncol[j] = nt * 64 + 16 * j + ln;
    brow[j] = WT + (size_t)ncol[j] * C_ + hi * 16;
  }
  for (int k0 = 0; k0 < C_; k0 += 32) {
    __builtin_prefetch((const void*)(arow0 + k0 + 32), 0, 0);
    AFrag a0, a1;
    a0.p.lo = *(const v8h*)(arow0 + k0);
    a0.p.hi = *(const v8h*)(arow0 + k0 + 8);
    a1.p.lo = *(const v8h*)(arow1 + k0);
    a1.p.hi = *(const v8h*)(arow1 + k0 + 8);
#pragma unroll
    for (int j = 0; j < 4; ++j) {
      v16h b = *(const v16h*)(brow[j] + k0);
      acc0[j] = WMMA_F16(a0.v, b, acc0[j]);
      acc1[j] = WMMA_F16(a1.v, b, acc1[j]);
    }
  }
#pragma unroll
  for (int j = 0; j < 4; ++j) {
    const float bias = b_proj[ncol[j]];
#pragma unroll
    for (int i = 0; i < 2; ++i) {
      const v8f* acc = (i == 0) ? acc0 : acc1;
#pragma unroll
      for (int r = 0; r < 8; ++r) {
        const int grow = mt * 32 + i * 16 + r + hi * 8;
        out[(size_t)grow * C_ + ncol[j]] = acc[r == r ? j : j][r] + bias;
      }
    }
  }
}

// ---------------------------------------------------------------------------
extern "C" void kernel_launch(void* const* d_in, const int* in_sizes, int n_in,
                              void* d_out, int out_size, void* d_ws, size_t ws_size,
                              hipStream_t stream) {
  const float* x      = (const float*)d_in[0];
  const float* ct     = (const float*)d_in[1];
  const float* W_attn = (const float*)d_in[2];
  const float* b_attn = (const float*)d_in[3];
  const float* W_proj = (const float*)d_in[4];
  const float* b_proj = (const float*)d_in[5];
  const float* lut    = (const float*)d_in[6];
  float* out = (float*)d_out;

  char* ws = (char*)d_ws;
  size_t off = 0;
  auto carve = [&](size_t bytes) {
    void* p = ws + off;
    off = (off + bytes + 255) & ~(size_t)255;
    return p;
  };
  _Float16* xh  = (_Float16*)carve((size_t)ROWS_ * C_ * 2);
  _Float16* WTa = (_Float16*)carve((size_t)NQKV_ * C_ * 2);
  _Float16* WTp = (_Float16*)carve((size_t)C_ * C_ * 2);
  _Float16* Qh  = (_Float16*)carve((size_t)B_ * H_ * T_ * HD_ * 2);
  _Float16* Kh  = (_Float16*)carve((size_t)B_ * H_ * T_ * HD_ * 2);
  _Float16* Vh  = (_Float16*)carve((size_t)B_ * H_ * T_ * HD_ * 2);
  _Float16* Yh  = (_Float16*)carve((size_t)ROWS_ * C_ * 2);

  // pre-pass conversions
  {
    int n = ROWS_ * C_;
    cvt_f16_kernel<<<(n + 255) / 256, 256, 0, stream>>>(x, xh, n);
    int na = C_ * NQKV_;
    transpose_f16_kernel<<<(na + 255) / 256, 256, 0, stream>>>(W_attn, WTa, C_, NQKV_);
    int np = C_ * C_;
    transpose_f16_kernel<<<(np + 255) / 256, 256, 0, stream>>>(W_proj, WTp, C_, C_);
  }
  // QKV projection: 250 * 24 waves (32x64 tiles), 8 waves/block
  gemm_qkv_kernel<<<(250 * 24) / 8, 256, 0, stream>>>(xh, WTa, b_attn, Qh, Kh, Vh);
  // fused attention: B*H*32 query tiles, 4 waves/block
  attn_kernel<<<(B_ * H_ * 32) / 4, 128, 0, stream>>>(Qh, Kh, Vh, ct, lut, Yh);
  // output projection: 250 * 8 waves (32x64 tiles), 8 waves/block
  gemm_proj_kernel<<<(250 * 8) / 8, 256, 0, stream>>>(Yh, WTp, b_proj, out);
}
